// NeuralPDA_77060303225020
// MI455X (gfx1250) — compile-verified, matrix operand order checked by reading, not run on hardware
//
#include <hip/hip_runtime.h>

// ---------------------------------------------------------------------------
// NeuralPDA decoder for MI455X (gfx1250, wave32, WMMA).
//  - Precompute Gx = emb@Wx+bx (all V rows) and Gh = tree_state@Wh+bh once
//    (GRU hidden is the loop-invariant tree_state in the reference), so the
//    10-step serial loop is gathers + two small GEMMs + reductions.
//  - All GEMMs use split-f16 ("f16x3"): x = hi + lo, C += hi*hi+hi*lo+lo*hi
//    via v_wmma_f32_16x16x32_f16, f32 accumulate (~22-bit mantissa).
//  - Register blocking: each wave computes a 16x64 strip (4 N-tiles),
//    reusing the A fragments -> 12 WMMAs per 10KB of loads per k-step.
//  - B (weight) buffers padded to 64 N-rows so the inner loop has no guards.
//  - exact_logit GEMM (2560x10000x512, 26 GFLOP) fuses the epilogue:
//    + btok + log(tok_table[symbol]+TINY), single pass over the 100MB output.
// ---------------------------------------------------------------------------

typedef __attribute__((ext_vector_type(16))) _Float16 v16h;
typedef __attribute__((ext_vector_type(8)))  _Float16 v8h;
typedef __attribute__((ext_vector_type(8)))  float    v8f;

#define TINYF 1e-13f

static constexpr int B_   = 256;
static constexpr int V_   = 2000;
static constexpr int VT_  = 10000;
static constexpr int D_   = 256;
static constexpr int OPT_ = 16;
static constexpr int SEQ_ = 10;
static constexpr int SRC_ = 40;

static inline int pad64(int n) { return (n + 63) & ~63; }

// ------------------------------ small utils -------------------------------

__global__ void k_i32_to_f32(const int* __restrict__ src, float* __restrict__ dst, int n) {
  int i = blockIdx.x * blockDim.x + threadIdx.x;
  if (i < n) dst[i] = (float)src[i];
}

// f32 -> f16 hi/lo split, row-major (for A matrices)
__global__ void k_split_row(const float* __restrict__ src,
                            _Float16* __restrict__ hi, _Float16* __restrict__ lo, int n) {
  int i = blockIdx.x * blockDim.x + threadIdx.x;
  if (i < n) {
    float x = src[i];
    _Float16 h = (_Float16)x;
    hi[i] = h;
    lo[i] = (_Float16)(x - (float)h);
  }
}

// f32 [K][N] -> f16 hi/lo transposed [N][K] (weights as B operand)
__global__ void k_split_T(const float* __restrict__ src,
                          _Float16* __restrict__ hiT, _Float16* __restrict__ loT,
                          int K, int N) {
  int i = blockIdx.x * blockDim.x + threadIdx.x;
  if (i < K * N) {
    int k = i / N, n = i % N;
    float x = src[i];
    _Float16 h = (_Float16)x;
    hiT[(size_t)n * K + k] = h;
    loT[(size_t)n * K + k] = (_Float16)(x - (float)h);
  }
}

__global__ void k_init(const float* __restrict__ tree_state,
                       float* __restrict__ ts, float* __restrict__ opts_logp) {
  int i = blockIdx.x * blockDim.x + threadIdx.x;
  if (i < B_ * D_) ts[i] = tree_state[i];
  if (i < B_ * OPT_) opts_logp[i] = 0.0f;
}

// ------------------------------ GEMM (WMMA) -------------------------------
// CDNA5 16-bit A-matrix 16x32 lane layout: lane<16 holds row M=lane,
// K = k0..k0+7 (elems 0..7) and k0+16..23 (elems 8..15); lanes 16..31 hold
// the same rows with K+8. B uses the mirrored layout on the pre-transposed
// [N][K] weight, so every fragment is two contiguous 16B loads.

__device__ __forceinline__ v16h load_frag(const _Float16* __restrict__ p, int k0) {
  v8h a = *(const v8h*)(p + k0);
  v8h b = *(const v8h*)(p + k0 + 16);
  return __builtin_shufflevector(a, b, 0, 1, 2, 3, 4, 5, 6, 7,
                                 8, 9, 10, 11, 12, 13, 14, 15);
}

#define WMMA3(cacc, ah_, al_, bf_, bl_)                                                             \
  cacc = __builtin_amdgcn_wmma_f32_16x16x32_f16(false, ah_, false, bf_, (short)0, cacc, false, false); \
  cacc = __builtin_amdgcn_wmma_f32_16x16x32_f16(false, ah_, false, bl_, (short)0, cacc, false, false); \
  cacc = __builtin_amdgcn_wmma_f32_16x16x32_f16(false, al_, false, bf_, (short)0, cacc, false, false);

// EPI: 0 = C = acc + bias
//      1 = soa path: C = tanh(acc+bias); also writes f16 hi/lo split of C and
//          updates ts = tanh(ts + C)   (unique element per lane -> race-free)
//      2 = exact_logit path: C = acc + bias + log(tok[sym[m]][n] + TINY)
// Each wave: one 16(M) x 64(N) strip; block = 4 waves = 16 x 256.
template <int EPI>
__global__ void k_gemm(const _Float16* __restrict__ Ah, const _Float16* __restrict__ Al,
                       const _Float16* __restrict__ Bh, const _Float16* __restrict__ Bl,
                       const float* __restrict__ bias, float* __restrict__ C,
                       int M, int N, int K,
                       _Float16* __restrict__ soaH, _Float16* __restrict__ soaL,
                       float* __restrict__ ts,
                       const float* __restrict__ tok, const int* __restrict__ sym) {
  const int wave = threadIdx.x >> 5;
  const int lane = threadIdx.x & 31;
  const int tn0 = (blockIdx.x * 4 + wave) * 64;    // 64-wide N strip per wave
  const int tm  = blockIdx.y * 16;                 // 16-tall M tile
  if (tn0 >= N) return;                            // wave-uniform exit
  const int half = lane >> 4;
  const int r16  = lane & 15;

  const _Float16* arowH = Ah + (size_t)(tm + r16) * K + half * 8;
  const _Float16* arowL = Al + (size_t)(tm + r16) * K + half * 8;
  const _Float16* b0H   = Bh + (size_t)(tn0 + r16) * K + half * 8;
  const _Float16* b0L   = Bl + (size_t)(tn0 + r16) * K + half * 8;
  const size_t tstr = (size_t)16 * K;              // row stride per 16-N tile

  v8f c0 = {}, c1 = {}, c2 = {}, c3 = {};
  for (int k0 = 0; k0 < K; k0 += 32) {
    __builtin_prefetch(b0H + k0 + 64, 0, 3);       // WGP-scope prefetch
    v16h ah = load_frag(arowH, k0);
    v16h al = load_frag(arowL, k0);
    {
      v16h bf = load_frag(b0H, k0);
      v16h bl = load_frag(b0L, k0);
      WMMA3(c0, ah, al, bf, bl)
    }
    {
      v16h bf = load_frag(b0H + tstr, k0);
      v16h bl = load_frag(b0L + tstr, k0);
      WMMA3(c1, ah, al, bf, bl)
    }
    {
      v16h bf = load_frag(b0H + 2 * tstr, k0);
      v16h bl = load_frag(b0L + 2 * tstr, k0);
      WMMA3(c2, ah, al, bf, bl)
    }
    {
      v16h bf = load_frag(b0H + 3 * tstr, k0);
      v16h bl = load_frag(b0L + 3 * tstr, k0);
      WMMA3(c3, ah, al, bf, bl)
    }
  }

  v8f cc[4] = {c0, c1, c2, c3};
#pragma unroll
  for (int j = 0; j < 4; j++) {
    const int tn = tn0 + j * 16;
    if (tn >= N) break;                            // Ns are multiples of 16
    const int n = tn + r16;
    const float bn = bias[n];
#pragma unroll
    for (int r = 0; r < 8; r++) {
      const int m = tm + r + half * 8;             // C layout: VGPR r -> M=r / r+8
      float acc = cc[j][r] + bn;
      size_t idx = (size_t)m * N + n;
      if (EPI == 0) {
        C[idx] = acc;
      } else if (EPI == 1) {
        float v = tanhf(acc);
        C[idx] = v;
        _Float16 vh = (_Float16)v;
        soaH[idx] = vh;
        soaL[idx] = (_Float16)(v - (float)vh);
        ts[idx] = tanhf(ts[idx] + v);
      } else {
        float cw = tok[(size_t)sym[m] * N + n];
        C[idx] = acc + logf(cw + TINYF);
      }
    }
  }
}

// ------------------------------ GRU gather --------------------------------

__global__ void k_gru(const int* __restrict__ inp, const float* __restrict__ Gx,
                      const float* __restrict__ Gh, const float* __restrict__ h0,
                      float* __restrict__ h) {
  const int b = blockIdx.x, d = threadIdx.x;
  const int row = inp[b];
  const float* gx = Gx + (size_t)row * (3 * D_);
  const float* gh = Gh + (size_t)b * (3 * D_);
  float r = 1.0f / (1.0f + expf(-(gx[d] + gh[d])));
  float z = 1.0f / (1.0f + expf(-(gx[D_ + d] + gh[D_ + d])));
  float n = tanhf(gx[2 * D_ + d] + r * gh[2 * D_ + d]);
  float hp = h0[(size_t)b * D_ + d];
  h[(size_t)b * D_ + d] = (1.0f - z) * n + z * hp;
}

// --------------------------- attention + concat ---------------------------

__global__ void k_attn(const float* __restrict__ h, const float* __restrict__ mem,
                       float* __restrict__ cat,
                       _Float16* __restrict__ catH, _Float16* __restrict__ catL) {
  __shared__ float sh[D_];
  __shared__ float ss[SRC_];
  const int b = blockIdx.x, t = threadIdx.x;
  sh[t] = h[(size_t)b * D_ + t];
  __syncthreads();

  const int wave = t >> 5, lane = t & 31;
  for (int j = wave; j < SRC_; j += 8) {
    const float* mrow = mem + ((size_t)b * SRC_ + j) * D_;
    float p = 0.0f;
    for (int d = lane; d < D_; d += 32) p += sh[d] * mrow[d];
    for (int o = 16; o > 0; o >>= 1) p += __shfl_xor(p, o, 32);
    if (lane == 0) ss[j] = p;
  }
  __syncthreads();
  if (t == 0) {                                    // 40-wide softmax: trivial
    float mx = ss[0];
    for (int j = 1; j < SRC_; j++) mx = fmaxf(mx, ss[j]);
    float sum = 0.0f;
    for (int j = 0; j < SRC_; j++) { float e = expf(ss[j] - mx); ss[j] = e; sum += e; }
    float inv = 1.0f / sum;
    for (int j = 0; j < SRC_; j++) ss[j] *= inv;
  }
  __syncthreads();
  float acc = 0.0f;
  for (int j = 0; j < SRC_; j++) acc += ss[j] * mem[((size_t)b * SRC_ + j) * D_ + t];

  const size_t i0 = (size_t)b * 2 * D_ + t, i1 = i0 + D_;
  const float hv = sh[t];
  cat[i0] = acc; cat[i1] = hv;
  _Float16 a16 = (_Float16)acc;
  catH[i0] = a16; catL[i0] = (_Float16)(acc - (float)a16);
  _Float16 h16 = (_Float16)hv;
  catH[i1] = h16; catL[i1] = (_Float16)(hv - (float)h16);
}

// ------------------- per-step mask / softmax-logp / argmax ----------------

__global__ void k_steppost(const float* __restrict__ logits, const int* __restrict__ gg,
                           int t, float* __restrict__ opts_logp, int* __restrict__ nxt) {
  __shared__ float s_cl[V_];
  __shared__ int   s_comp[OPT_];
  __shared__ int   s_zero;
  __shared__ float s_red[256];
  __shared__ int   s_idx[256];
  const int b = blockIdx.x, tid = threadIdx.x;
  if (tid == 0) s_zero = 0;
  __syncthreads();
  if (tid < OPT_) {
    int sy = gg[b * OPT_ * 4 * SEQ_ + tid * 4 * SEQ_ + 0 * SEQ_ + t];
    int mk = gg[b * OPT_ * 4 * SEQ_ + tid * 4 * SEQ_ + 3 * SEQ_ + t];
    s_comp[tid] = sy * mk;
    if (mk != 0 && sy * mk == 0) atomicOr(&s_zero, 1);
  }
  __syncthreads();

  const float LOG_T = logf(TINYF);
  const float LOG_1 = logf(1.0f + TINYF);
  float lmax = -3.4e38f; int limax = 0;
  for (int v = tid; v < V_; v += 256) {
    bool allowed;
    if (v == 0) {
      allowed = (s_zero != 0);
    } else {
      allowed = false;
#pragma unroll
      for (int o = 0; o < OPT_; o++) allowed = allowed || (s_comp[o] == v);
    }
    float cl = logits[(size_t)b * V_ + v] + (allowed ? LOG_1 : LOG_T);
    s_cl[v] = cl;
    if (cl > lmax) { lmax = cl; limax = v; }
  }
  s_red[tid] = lmax; s_idx[tid] = limax;
  __syncthreads();
  for (int s = 128; s > 0; s >>= 1) {
    if (tid < s) {
      float ov = s_red[tid + s]; int oi = s_idx[tid + s];
      if (ov > s_red[tid] || (ov == s_red[tid] && oi < s_idx[tid])) {
        s_red[tid] = ov; s_idx[tid] = oi;
      }
    }
    __syncthreads();
  }
  const float gmax = s_red[0];
  const int garg = s_idx[0];
  __syncthreads();

  float psum = 0.0f;
  for (int v = tid; v < V_; v += 256) psum += expf(s_cl[v] - gmax);
  s_red[tid] = psum;
  __syncthreads();
  for (int s = 128; s > 0; s >>= 1) {
    if (tid < s) s_red[tid] += s_red[tid + s];
    __syncthreads();
  }
  const float sum = s_red[0];

  if (tid == 0) nxt[b] = garg;
  if (tid < OPT_) {
    int sy = gg[b * OPT_ * 4 * SEQ_ + tid * 4 * SEQ_ + 0 * SEQ_ + t];
    int mk = gg[b * OPT_ * 4 * SEQ_ + tid * 4 * SEQ_ + 3 * SEQ_ + t];
    float p = expf(s_cl[sy] - gmax) / sum;                 // softmax value
    opts_logp[b * OPT_ + tid] += logf(p + TINYF) * (float)mk;
  }
}

// ----------------------- best-option select + topo ------------------------

__global__ void k_select(const float* __restrict__ opts_logp, const int* __restrict__ gg,
                         float* __restrict__ out_opts, float* __restrict__ out_sym,
                         float* __restrict__ out_pg, float* __restrict__ out_fg,
                         float* __restrict__ out_mk, int* __restrict__ sym_i) {
  const int b = blockIdx.x, t = threadIdx.x;
  __shared__ int s_best;
  if (t == 0) {
    float bv = opts_logp[b * OPT_]; int bi = 0;
    for (int o = 1; o < OPT_; o++) {
      float v = opts_logp[b * OPT_ + o];
      if (v > bv) { bv = v; bi = o; }
    }
    s_best = bi;
  }
  __syncthreads();
  if (t < OPT_) out_opts[b * OPT_ + t] = opts_logp[b * OPT_ + t];
  const int* base = gg + b * OPT_ * 4 * SEQ_ + s_best * 4 * SEQ_;
  if (t < SEQ_) {
    int sy = base[0 * SEQ_ + t];
    out_sym[b * SEQ_ + t] = (float)sy;
    out_pg[b * SEQ_ + t]  = (float)base[1 * SEQ_ + t];
    out_fg[b * SEQ_ + t]  = (float)base[2 * SEQ_ + t];
    out_mk[b * SEQ_ + t]  = (float)base[3 * SEQ_ + t];
    sym_i[b * SEQ_ + t] = sy;
  }
}

// ------------------ build A = [emb[sym], ts_final] (hi/lo) ----------------

__global__ void k_projA(const int* __restrict__ sym_i, const float* __restrict__ emb,
                        const float* __restrict__ ts,
                        _Float16* __restrict__ AH, _Float16* __restrict__ AL) {
  const int m = blockIdx.x, t = threadIdx.x;
  const int b = m / SEQ_;
  const int sy = sym_i[m];
  float x0 = emb[(size_t)sy * D_ + t];
  float x1 = ts[(size_t)b * D_ + t];
  size_t i0 = (size_t)m * (2 * D_) + t, i1 = i0 + D_;
  _Float16 h0 = (_Float16)x0; AH[i0] = h0; AL[i0] = (_Float16)(x0 - (float)h0);
  _Float16 h1 = (_Float16)x1; AH[i1] = h1; AL[i1] = (_Float16)(x1 - (float)h1);
}

// --------------------------------- host -----------------------------------

extern "C" void kernel_launch(void* const* d_in, const int* in_sizes, int n_in,
                              void* d_out, int out_size, void* d_ws, size_t ws_size,
                              hipStream_t stream) {
  (void)in_sizes; (void)n_in; (void)out_size; (void)ws_size;

  const int*   lhs        = (const int*)d_in[0];
  const int*   lhs_mask   = (const int*)d_in[1];
  const float* tree_state = (const float*)d_in[2];
  const int*   gg         = (const int*)d_in[3];
  const float* emb        = (const float*)d_in[4];
  const float* mem        = (const float*)d_in[5];
  const float* Wx         = (const float*)d_in[6];
  const float* Wh         = (const float*)d_in[7];
  const float* bx         = (const float*)d_in[8];
  const float* bhp        = (const float*)d_in[9];
  const float* Wc         = (const float*)d_in[10];
  const float* bc         = (const float*)d_in[11];
  const float* Wsym       = (const float*)d_in[12];
  const float* bsym       = (const float*)d_in[13];
  const float* Wtok       = (const float*)d_in[14];
  const float* btok       = (const float*)d_in[15];
  const float* tok_table  = (const float*)d_in[16];
  float* out = (float*)d_out;

  const int Vp  = pad64(V_);    // 2048 padded N-rows for WsymT
  const int VTp = pad64(VT_);   // 10048 padded N-rows for WtokT

  // workspace carve-out (256B aligned)
  char* wsb = (char*)d_ws;
  size_t off = 0;
  auto alloc = [&](size_t bytes) -> void* {
    void* p = wsb + off;
    off += (bytes + 255) & ~(size_t)255;
    return p;
  };
  _Float16* embH  = (_Float16*)alloc((size_t)V_ * D_ * 2);
  _Float16* embL  = (_Float16*)alloc((size_t)V_ * D_ * 2);
  _Float16* ts0H  = (_Float16*)alloc((size_t)B_ * D_ * 2);
  _Float16* ts0L  = (_Float16*)alloc((size_t)B_ * D_ * 2);
  _Float16* WxTH  = (_Float16*)alloc((size_t)D_ * 3 * D_ * 2);
  _Float16* WxTL  = (_Float16*)alloc((size_t)D_ * 3 * D_ * 2);
  _Float16* WhTH  = (_Float16*)alloc((size_t)D_ * 3 * D_ * 2);
  _Float16* WhTL  = (_Float16*)alloc((size_t)D_ * 3 * D_ * 2);
  _Float16* WcTH  = (_Float16*)alloc((size_t)2 * D_ * D_ * 2);
  _Float16* WcTL  = (_Float16*)alloc((size_t)2 * D_ * D_ * 2);
  _Float16* WsTH  = (_Float16*)alloc((size_t)D_ * Vp * 2);   // padded rows
  _Float16* WsTL  = (_Float16*)alloc((size_t)D_ * Vp * 2);
  _Float16* WtTH  = (_Float16*)alloc((size_t)2 * D_ * VTp * 2);
  _Float16* WtTL  = (_Float16*)alloc((size_t)2 * D_ * VTp * 2);
  float* Gx     = (float*)alloc((size_t)V_ * 3 * D_ * 4);
  float* Gh     = (float*)alloc((size_t)B_ * 3 * D_ * 4);
  float* hbuf   = (float*)alloc((size_t)B_ * D_ * 4);
  float* cat    = (float*)alloc((size_t)B_ * 2 * D_ * 4);
  _Float16* catH = (_Float16*)alloc((size_t)B_ * 2 * D_ * 2);
  _Float16* catL = (_Float16*)alloc((size_t)B_ * 2 * D_ * 2);
  float* soa    = (float*)alloc((size_t)B_ * D_ * 4);
  _Float16* soaH = (_Float16*)alloc((size_t)B_ * D_ * 2);
  _Float16* soaL = (_Float16*)alloc((size_t)B_ * D_ * 2);
  float* tsbuf  = (float*)alloc((size_t)B_ * D_ * 4);
  float* logits = (float*)alloc((size_t)B_ * V_ * 4);
  float* optsl  = (float*)alloc((size_t)B_ * OPT_ * 4);
  int*   nxt    = (int*)alloc((size_t)B_ * 4);
  int*   sym_i  = (int*)alloc((size_t)B_ * SEQ_ * 4);
  _Float16* pAH = (_Float16*)alloc((size_t)B_ * SEQ_ * 2 * D_ * 2);
  _Float16* pAL = (_Float16*)alloc((size_t)B_ * SEQ_ * 2 * D_ * 2);

  // output layout (flat, reference return order)
  float* o_lhs  = out;
  float* o_mask = out + 256;
  float* o_gg   = out + 512;
  float* o_opts = o_gg + B_ * OPT_ * 4 * SEQ_;
  float* o_sym  = o_opts + B_ * OPT_;
  float* o_pg   = o_sym + B_ * SEQ_;
  float* o_fg   = o_pg + B_ * SEQ_;
  float* o_mk   = o_fg + B_ * SEQ_;
  float* o_ex   = o_mk + B_ * SEQ_;

  // pass-through outputs (int -> float)
  k_i32_to_f32<<<1, 256, 0, stream>>>(lhs, o_lhs, B_);
  k_i32_to_f32<<<1, 256, 0, stream>>>(lhs_mask, o_mask, B_);
  k_i32_to_f32<<<(B_ * OPT_ * 4 * SEQ_ + 255) / 256, 256, 0, stream>>>(gg, o_gg, B_ * OPT_ * 4 * SEQ_);

  // precision splits + weight transposes
  k_split_row<<<(V_ * D_ + 255) / 256, 256, 0, stream>>>(emb, embH, embL, V_ * D_);
  k_split_row<<<(B_ * D_ + 255) / 256, 256, 0, stream>>>(tree_state, ts0H, ts0L, B_ * D_);
  k_split_T<<<(D_ * 3 * D_ + 255) / 256, 256, 0, stream>>>(Wx, WxTH, WxTL, D_, 3 * D_);
  k_split_T<<<(D_ * 3 * D_ + 255) / 256, 256, 0, stream>>>(Wh, WhTH, WhTL, D_, 3 * D_);
  k_split_T<<<(2 * D_ * D_ + 255) / 256, 256, 0, stream>>>(Wc, WcTH, WcTL, 2 * D_, D_);
  k_split_T<<<(D_ * V_ + 255) / 256, 256, 0, stream>>>(Wsym, WsTH, WsTL, D_, V_);
  k_split_T<<<(2 * D_ * VT_ + 255) / 256, 256, 0, stream>>>(Wtok, WtTH, WtTL, 2 * D_, VT_);
  k_init<<<(B_ * D_ + 255) / 256, 256, 0, stream>>>(tree_state, tsbuf, optsl);

  const dim3 blk(128);  // 4 waves, each a 16x64 strip -> block covers 16x256
  // Gx = emb @ Wx + bx   (2000 x 768 x 256)
  k_gemm<0><<<dim3((3 * D_ + 255) / 256, V_ / 16), blk, 0, stream>>>(
      embH, embL, WxTH, WxTL, bx, Gx, V_, 3 * D_, D_,
      nullptr, nullptr, nullptr, nullptr, nullptr);
  // Gh = tree_state @ Wh + bh   (256 x 768 x 256) — loop-invariant
  k_gemm<0><<<dim3((3 * D_ + 255) / 256, B_ / 16), blk, 0, stream>>>(
      ts0H, ts0L, WhTH, WhTL, bhp, Gh, B_, 3 * D_, D_,
      nullptr, nullptr, nullptr, nullptr, nullptr);

  // serial decode loop (argmax feeds next step's embedding row)
  for (int t = 0; t < SEQ_; t++) {
    const int* inp = (t == 0) ? lhs : (const int*)nxt;
    k_gru<<<B_, D_, 0, stream>>>(inp, Gx, Gh, tree_state, hbuf);
    k_attn<<<B_, D_, 0, stream>>>(hbuf, mem, cat, catH, catL);
    // soa = tanh([ctx,h]@Wc + bc); fused ts = tanh(ts + soa)
    k_gemm<1><<<dim3((D_ + 255) / 256, B_ / 16), blk, 0, stream>>>(
        catH, catL, WcTH, WcTL, bc, soa, B_, D_, 2 * D_,
        soaH, soaL, tsbuf, nullptr, nullptr);
    // logits = soa @ Wsym + bsym   (256 x 2000 x 256)
    k_gemm<0><<<dim3((V_ + 255) / 256, B_ / 16), blk, 0, stream>>>(
        soaH, soaL, WsTH, WsTL, bsym, logits, B_, V_, D_,
        nullptr, nullptr, nullptr, nullptr, nullptr);
    k_steppost<<<B_, 256, 0, stream>>>(logits, gg, t, optsl, nxt);
  }

  k_select<<<B_, 32, 0, stream>>>(optsl, gg, o_opts, o_sym, o_pg, o_fg, o_mk, sym_i);
  k_projA<<<B_ * SEQ_, D_, 0, stream>>>(sym_i, emb, tsbuf, pAH, pAL);
  // exact_logit = proj @ Wtok + btok + log(tok_table[sym] + TINY)
  // (2560 x 10000 x 512) — the 26-GFLOP GEMM, fused epilogue
  k_gemm<2><<<dim3((VT_ + 255) / 256, (B_ * SEQ_) / 16), blk, 0, stream>>>(
      pAH, pAL, WtTH, WtTL, btok, o_ex, B_ * SEQ_, VT_, 2 * D_,
      nullptr, nullptr, nullptr, tok_table, sym_i);
}
